// IMPContext_11776800326171
// MI455X (gfx1250) — matrix operand order repaired and checked by reading, not compiled, not run on hardware
//
#include <hip/hip_runtime.h>
#include <math.h>

// ---------------------------------------------------------------------------
// Problem constants (match reference)
// ---------------------------------------------------------------------------
#define H       512
#define IN_DIM  4096
#define POOL    4096
#define N_OBJ   8192
#define N_REL   65536
#define OBJ_CLS 151
#define REL_CLS 51
#define NUM_ITER 3
#define CHUNK   16384           // edge-GRU row chunk (N_REL / 4)

// ---------------------------------------------------------------------------
// WMMA vector types (gfx1250, wave32)
// ---------------------------------------------------------------------------
typedef __attribute__((ext_vector_type(16))) __bf16        bf16x16;
typedef __attribute__((ext_vector_type(8)))  float         floatx8;
typedef __attribute__((ext_vector_type(8)))  unsigned int  uintx8;

#define LDS_STRIDE 40   // 32 K-halves + 8 pad -> 20-bank row stride, conflict-free

// f32 -> bf16 round-to-nearest-even, pure integer (always lowers)
__device__ __forceinline__ unsigned short f2bf(float f) {
    unsigned int u = __builtin_bit_cast(unsigned int, f);
    u += 0x7fffu + ((u >> 16) & 1u);
    return (unsigned short)(u >> 16);
}

// A fragment: 16x32 bf16, ISA layout:
//   lanes 0-15  (M=lane):    V0-3 K=0..7,  V4-7 K=16..23
//   lanes 16-31 (M=lane-16): V0-3 K=8..15, V4-7 K=24..31
__device__ __forceinline__ bf16x16 load_frag_a(const unsigned short* tile, int mBase, int lane) {
    int row = mBase + (lane & 15);
    int kh  = (lane >> 4) << 3;
    uintx8 t;
#pragma unroll
    for (int i = 0; i < 8; ++i) {
        int kb = ((i < 4) ? (2 * i) : (16 + 2 * (i - 4))) + kh;
        t[i] = *(const unsigned int*)&tile[row * LDS_STRIDE + kb];
    }
    return __builtin_bit_cast(bf16x16, t);
}

// B fragment: 32x16 bf16 (K x N). N = lane&15; lanes 0-15 K=0..7,16..23, lanes 16-31 K=8..15,24..31.
// LDS tile is stored N-major: tile[n*LDS_STRIDE + k]  (== rows of W, coalesced load)
__device__ __forceinline__ bf16x16 load_frag_b(const unsigned short* tile, int nBase, int lane) {
    int nrow = nBase + (lane & 15);
    int kOff = (lane >> 4) << 3;
    uintx8 t;
#pragma unroll
    for (int i = 0; i < 8; ++i)
        t[i] = *(const unsigned int*)&tile[nrow * LDS_STRIDE + kOff + ((i < 4) ? (2 * i) : (16 + 2 * (i - 4)))];
    return __builtin_bit_cast(bf16x16, t);
}

// C/D 16x16 f32 store: VGPR r -> M = r + 8*(lane>=16); N = lane&15
__device__ __forceinline__ void store_tile(const floatx8& acc, float* __restrict__ C,
                                           const float* __restrict__ bias,
                                           int M, int N, int relu,
                                           int mBase, int nBase, int lane) {
    int col = nBase + (lane & 15);
    if (col >= N) return;
    int rtop = mBase + ((lane >> 4) << 3);
    float b = bias[col];
#pragma unroll
    for (int r = 0; r < 8; ++r) {
        int row = rtop + r;
        if (row < M) {
            float v = acc[r] + b;
            if (relu) v = v > 0.f ? v : 0.f;
            C[(size_t)row * N + col] = v;
        }
    }
}

// ---------------------------------------------------------------------------
// GEMM:  C[M,N] = A[M,K] * W[N,K]^T + bias[N]   (optional ReLU)
// block = 256 threads (8 waves); block tile 128(M) x 128(N);
// wave tile 32(M) x 64(N) -> 8 v_wmma per K-step (K stepped by 32).
// Double-buffered LDS software pipeline: one barrier per K-step, global
// fetch of tile k+1 overlaps WMMA on tile k.
// FULL=true: M,N multiples of 128 -> no guards in the hot loop.
// ---------------------------------------------------------------------------
template<bool FULL>
__global__ __launch_bounds__(256)
void gemm_bf16_wmma(const float* __restrict__ A, const float* __restrict__ W,
                    const float* __restrict__ bias, float* __restrict__ C,
                    int M, int N, int K, int relu) {
    __shared__ unsigned short lsA[2][128 * LDS_STRIDE];
    __shared__ unsigned short lsB[2][128 * LDS_STRIDE];

    const int tid  = threadIdx.x;
    const int lane = tid & 31;
    const int wave = tid >> 5;
    const int blockM = blockIdx.y * 128;
    const int blockN = blockIdx.x * 128;
    const int wm = (wave >> 1) * 32;  // 0,32,64,96
    const int wn = (wave & 1) * 64;   // 0,64

    floatx8 acc[2][4];
#pragma unroll
    for (int i = 0; i < 2; ++i)
#pragma unroll
        for (int j = 0; j < 4; ++j) acc[i][j] = (floatx8){};

    // staging assignment: both tiles are 128 rows x 32 floats
    const int srow = tid >> 1;        // 0..127
    const int scol = (tid & 1) * 16;  // 16 floats per thread
    const bool okA = FULL || (blockM + srow < M);
    const bool okB = FULL || (blockN + srow < N);
    const float* srcA = A + (size_t)(blockM + srow) * K + scol;
    const float* srcB = W + (size_t)(blockN + srow) * K + scol;

    float ra[16], rb[16];

    auto fetch = [&](int k0) {
        if (okA) {
#pragma unroll
            for (int i = 0; i < 16; i += 4) {
                float4 v = *(const float4*)(srcA + k0 + i);
                ra[i] = v.x; ra[i + 1] = v.y; ra[i + 2] = v.z; ra[i + 3] = v.w;
            }
        } else {
#pragma unroll
            for (int i = 0; i < 16; ++i) ra[i] = 0.f;
        }
        if (okB) {
#pragma unroll
            for (int i = 0; i < 16; i += 4) {
                float4 v = *(const float4*)(srcB + k0 + i);
                rb[i] = v.x; rb[i + 1] = v.y; rb[i + 2] = v.z; rb[i + 3] = v.w;
            }
        } else {
#pragma unroll
            for (int i = 0; i < 16; ++i) rb[i] = 0.f;
        }
    };
    auto commit = [&](int p) {
        unsigned short* da = &lsA[p][srow * LDS_STRIDE + scol];
        unsigned short* db = &lsB[p][srow * LDS_STRIDE + scol];
#pragma unroll
        for (int i = 0; i < 16; ++i) da[i] = f2bf(ra[i]);
#pragma unroll
        for (int i = 0; i < 16; ++i) db[i] = f2bf(rb[i]);
    };

    // pipeline prologue
    fetch(0);
    commit(0);
    __syncthreads();

    int p = 0;
    for (int k0 = 0; k0 < K; k0 += 32) {
        const bool more = (k0 + 32) < K;
        if (more) fetch(k0 + 32);          // global loads overlap WMMA below

        bf16x16 af[2], bfr[4];
#pragma unroll
        for (int i = 0; i < 2; ++i) af[i] = load_frag_a(lsA[p], wm + 16 * i, lane);
#pragma unroll
        for (int j = 0; j < 4; ++j) bfr[j] = load_frag_b(lsB[p], wn + 16 * j, lane);

#pragma unroll
        for (int i = 0; i < 2; ++i)
#pragma unroll
            for (int j = 0; j < 4; ++j)
                acc[i][j] = __builtin_amdgcn_wmma_f32_16x16x32_bf16(
                    false, af[i], false, bfr[j], (short)0, acc[i][j], false, false);

        if (more) commit(p ^ 1);           // write other buffer: no read/write conflict
        __syncthreads();                   // single barrier per K-step
        p ^= 1;
    }

#pragma unroll
    for (int i = 0; i < 2; ++i)
#pragma unroll
        for (int j = 0; j < 4; ++j)
            store_tile(acc[i][j], C, bias, M, N, relu,
                       blockM + wm + 16 * i, blockN + wn + 16 * j, lane);
}

// ---------------------------------------------------------------------------
// GRU combine: h' = (1-z)*n + z*h  from gi = x@Wih^T+bih, gh = h@Whh^T+bhh
// hzero==1: h==0 -> gh collapses to b_hh (gh GEMM skipped by caller)
// ---------------------------------------------------------------------------
__global__ __launch_bounds__(256)
void gru_combine(const float* __restrict__ gi, const float* __restrict__ gh,
                 const float* __restrict__ h, const float* __restrict__ b_hh,
                 float* __restrict__ out, int rows, int hzero) {
    int idx = blockIdx.x * blockDim.x + threadIdx.x;
    if (idx >= rows * H) return;
    int row = idx >> 9;          // /512
    int col = idx & (H - 1);
    size_t gbase = (size_t)row * (3 * H) + col;
    float ir = gi[gbase], iz = gi[gbase + H], inn = gi[gbase + 2 * H];
    float hr, hz, hn, hv;
    if (hzero) {
        hr = b_hh[col]; hz = b_hh[col + H]; hn = b_hh[col + 2 * H]; hv = 0.f;
    } else {
        hr = gh[gbase]; hz = gh[gbase + H]; hn = gh[gbase + 2 * H];
        hv = h[(size_t)row * H + col];
    }
    float r = 1.f / (1.f + __expf(-(ir + hr)));
    float z = 1.f / (1.f + __expf(-(iz + hz)));
    float n = tanhf(inn + r * hn);
    out[(size_t)row * H + col] = (1.f - z) * n + z * hv;
}

// ---------------------------------------------------------------------------
// Fused gates + message + scatter: one wave (32 lanes) per relation.
//   g_* = sigmoid([a, edge] @ w_* + b_*)
//   msg            = g_sub*sub_v + g_obj*obj_v
//   vert_ctx[sub] += g_out*edge ; vert_ctx[obj] += g_in*edge   (atomics)
// ---------------------------------------------------------------------------
__global__ __launch_bounds__(256)
void gates_kernel(const float* __restrict__ vert, const float* __restrict__ edge,
                  const int* __restrict__ sub_idx, const int* __restrict__ obj_idx,
                  const float* __restrict__ w_sub, const float* __restrict__ b_sub,
                  const float* __restrict__ w_obj, const float* __restrict__ b_obj,
                  const float* __restrict__ w_out, const float* __restrict__ b_out,
                  const float* __restrict__ w_in,  const float* __restrict__ b_in,
                  float* __restrict__ msg, float* __restrict__ vctx, int n_rel) {
    int rel  = (int)((blockIdx.x * blockDim.x + threadIdx.x) >> 5);
    int lane = threadIdx.x & 31;
    if (rel >= n_rel) return;
    int si = sub_idx[rel], oi = obj_idx[rel];
    const float* sv_p = vert + (size_t)si * H;
    const float* ov_p = vert + (size_t)oi * H;
    const float* ev_p = edge + (size_t)rel * H;

    float sv[16], ov[16], ev[16];
    float d_sub = 0.f, d_obj = 0.f, d_out = 0.f, d_in = 0.f;
#pragma unroll
    for (int j = 0; j < 16; ++j) {
        int c = lane + 32 * j;
        float s = sv_p[c], o = ov_p[c], e = ev_p[c];
        sv[j] = s; ov[j] = o; ev[j] = e;
        d_sub += w_sub[c] * s + w_sub[H + c] * e;
        d_obj += w_obj[c] * o + w_obj[H + c] * e;
        d_out += w_out[c] * s + w_out[H + c] * e;
        d_in  += w_in[c]  * o + w_in[H + c]  * e;
    }
#pragma unroll
    for (int off = 16; off > 0; off >>= 1) {
        d_sub += __shfl_xor(d_sub, off, 32);
        d_obj += __shfl_xor(d_obj, off, 32);
        d_out += __shfl_xor(d_out, off, 32);
        d_in  += __shfl_xor(d_in,  off, 32);
    }
    float g_sub = 1.f / (1.f + __expf(-(d_sub + b_sub[0])));
    float g_obj = 1.f / (1.f + __expf(-(d_obj + b_obj[0])));
    float g_out = 1.f / (1.f + __expf(-(d_out + b_out[0])));
    float g_in  = 1.f / (1.f + __expf(-(d_in  + b_in[0])));

    float* m_p = msg + (size_t)rel * H;
#pragma unroll
    for (int j = 0; j < 16; ++j) {
        int c = lane + 32 * j;
        m_p[c] = g_sub * sv[j] + g_obj * ov[j];
        atomicAdd(&vctx[(size_t)si * H + c], g_out * ev[j]);
        atomicAdd(&vctx[(size_t)oi * H + c], g_in  * ev[j]);
    }
}

// ---------------------------------------------------------------------------
// Host-side helpers
// ---------------------------------------------------------------------------
static inline void launch_gemm(const float* A, const float* W, const float* bias,
                               float* C, int M, int N, int K, int relu, hipStream_t s) {
    dim3 grid((N + 127) / 128, (M + 127) / 128);
    if ((M % 128 == 0) && (N % 128 == 0))
        gemm_bf16_wmma<true><<<grid, 256, 0, s>>>(A, W, bias, C, M, N, K, relu);
    else
        gemm_bf16_wmma<false><<<grid, 256, 0, s>>>(A, W, bias, C, M, N, K, relu);
}

static inline void launch_combine(const float* gi, const float* gh, const float* h,
                                  const float* b_hh, float* out, int rows, int hzero,
                                  hipStream_t s) {
    int total = rows * H;
    gru_combine<<<(total + 255) / 256, 256, 0, s>>>(gi, gh, h, b_hh, out, rows, hzero);
}

extern "C" void kernel_launch(void* const* d_in, const int* in_sizes, int n_in,
                              void* d_out, int out_size, void* d_ws, size_t ws_size,
                              hipStream_t stream) {
    const float* x            = (const float*)d_in[0];
    const float* uf           = (const float*)d_in[1];
    const int*   sub_idx      = (const int*)d_in[2];
    const int*   obj_idx      = (const int*)d_in[3];
    const float* w_obj_unary  = (const float*)d_in[4];
    const float* b_obj_unary  = (const float*)d_in[5];
    const float* w_edge_unary = (const float*)d_in[6];
    const float* b_edge_unary = (const float*)d_in[7];
    const float* node_w_ih    = (const float*)d_in[8];
    const float* node_w_hh    = (const float*)d_in[9];
    const float* node_b_ih    = (const float*)d_in[10];
    const float* node_b_hh    = (const float*)d_in[11];
    const float* edge_w_ih    = (const float*)d_in[12];
    const float* edge_w_hh    = (const float*)d_in[13];
    const float* edge_b_ih    = (const float*)d_in[14];
    const float* edge_b_hh    = (const float*)d_in[15];
    const float* w_sub_g      = (const float*)d_in[16];
    const float* b_sub_g      = (const float*)d_in[17];
    const float* w_obj_g      = (const float*)d_in[18];
    const float* b_obj_g      = (const float*)d_in[19];
    const float* w_out_g      = (const float*)d_in[20];
    const float* b_out_g      = (const float*)d_in[21];
    const float* w_in_g       = (const float*)d_in[22];
    const float* b_in_g       = (const float*)d_in[23];
    const float* w_obj_fc     = (const float*)d_in[24];
    const float* b_obj_fc     = (const float*)d_in[25];
    const float* w_rel_fc     = (const float*)d_in[26];
    const float* b_rel_fc     = (const float*)d_in[27];

    float* obj_out = (float*)d_out;                       // [N_OBJ, OBJ_CLS]
    float* rel_out = obj_out + (size_t)N_OBJ * OBJ_CLS;   // [N_REL, REL_CLS]

    // ---- workspace layout (floats) ----
    float* ws = (float*)d_ws;
    size_t o = 0;
    float* vertA = ws + o; o += (size_t)N_OBJ * H;
    float* vertB = ws + o; o += (size_t)N_OBJ * H;
    float* vctx  = ws + o; o += (size_t)N_OBJ * H;        // also obj_rep staging
    float* edgeA = ws + o; o += (size_t)N_REL * H;
    float* edgeB = ws + o; o += (size_t)N_REL * H;
    float* msg   = ws + o; o += (size_t)N_REL * H;        // also rel_rep staging
    float* gi    = ws + o; o += (size_t)CHUNK * 3 * H;
    float* gh    = ws + o; o += (size_t)CHUNK * 3 * H;
    (void)ws_size; (void)in_sizes; (void)n_in; (void)out_size;

    // ---- unary projections ----
    launch_gemm(x,  w_obj_unary,  b_obj_unary,  vctx, N_OBJ, H, IN_DIM, 0, stream); // obj_rep
    launch_gemm(uf, w_edge_unary, b_edge_unary, msg,  N_REL, H, POOL,   1, stream); // rel_rep (ReLU)

    // ---- initial GRU states (h == 0 -> skip gh GEMM) ----
    launch_gemm(vctx, node_w_ih, node_b_ih, gi, N_OBJ, 3 * H, H, 0, stream);
    launch_combine(gi, nullptr, nullptr, node_b_hh, vertA, N_OBJ, 1, stream);
    for (int c = 0; c < N_REL / CHUNK; ++c) {
        size_t off = (size_t)c * CHUNK * H;
        launch_gemm(msg + off, edge_w_ih, edge_b_ih, gi, CHUNK, 3 * H, H, 0, stream);
        launch_combine(gi, nullptr, nullptr, edge_b_hh, edgeA + off, CHUNK, 1, stream);
    }

    // ---- message passing iterations ----
    for (int it = 0; it < NUM_ITER; ++it) {
        hipMemsetAsync(vctx, 0, (size_t)N_OBJ * H * sizeof(float), stream);
        gates_kernel<<<(N_REL * 32 + 255) / 256, 256, 0, stream>>>(
            vertA, edgeA, sub_idx, obj_idx,
            w_sub_g, b_sub_g, w_obj_g, b_obj_g,
            w_out_g, b_out_g, w_in_g, b_in_g,
            msg, vctx, N_REL);

        // edge GRU (chunked): edgeB = GRU(msg, edgeA)
        for (int c = 0; c < N_REL / CHUNK; ++c) {
            size_t off = (size_t)c * CHUNK * H;
            launch_gemm(msg + off,   edge_w_ih, edge_b_ih, gi, CHUNK, 3 * H, H, 0, stream);
            launch_gemm(edgeA + off, edge_w_hh, edge_b_hh, gh, CHUNK, 3 * H, H, 0, stream);
            launch_combine(gi, gh, edgeA + off, edge_b_hh, edgeB + off, CHUNK, 0, stream);
        }
        // node GRU: vertB = GRU(vert_ctx, vertA)
        launch_gemm(vctx,  node_w_ih, node_b_ih, gi, N_OBJ, 3 * H, H, 0, stream);
        launch_gemm(vertA, node_w_hh, node_b_hh, gh, N_OBJ, 3 * H, H, 0, stream);
        launch_combine(gi, gh, vertA, node_b_hh, vertB, N_OBJ, 0, stream);

        float* t;
        t = vertA; vertA = vertB; vertB = t;
        t = edgeA; edgeA = edgeB; edgeB = t;
    }

    // ---- output heads ----
    launch_gemm(vertA, w_obj_fc, b_obj_fc, obj_out, N_OBJ, OBJ_CLS, H, 0, stream);
    launch_gemm(edgeA, w_rel_fc, b_rel_fc, rel_out, N_REL, REL_CLS, H, 0, stream);
}
